// PointNet2MSGSeg_31542239822574
// MI455X (gfx1250) — matrix operand order repaired and checked by reading, not compiled, not run on hardware
//
#include <hip/hip_runtime.h>
#include <hip/hip_bf16.h>

// ---------------------------------------------------------------------------
// PointNet++ MSG segmentation forward pass for gfx1250 (MI455X).
// All Linear(+ReLU) layers run through a wave32 WMMA GEMM
// (v_wmma_f32_16x16x32_f16, f16 in / f32 accumulate). Activations are kept in
// f16 with channel dims padded to multiples of 32 (pad lanes stay exactly 0
// through bias+ReLU). Irregular stages (FPS, ball query, kNN, pooling,
// interpolation, log-softmax) are scalar/VALU kernels.
// ---------------------------------------------------------------------------

#define BATCH 8
#define NPTS  4096
#define SS1   512
#define SS2   128

typedef __attribute__((ext_vector_type(16))) _Float16 v16h;
typedef __attribute__((ext_vector_type(8)))  _Float16 v8h;
typedef __attribute__((ext_vector_type(8)))  float    v8f;

union Frag16 { v16h v; v8h h[2]; };

// ---------------------------------------------------------------------------
// Generic WMMA GEMM: Y[M,Npad](f16) = act(X[M,Kpad](f16) @ Wt^T + bias)
// Wt is stored [Npad][Kpad] (i.e. W transposed), bias is f32 [Npad] (pad=0).
// One wave computes a 32x32 output tile (2 A frags x 2 B frags -> 4 WMMAs per
// 32-wide K step = 2 b128 loads per WMMA); block = 4 waves = 128 rows.
// Requires M % 32 == 0 (guaranteed by construction here).
// Fragment layouts per CDNA5 ISA 7.12.2:
//   A (16x32 f16): lane&15 = row M; lane>>4 selects K {0..7,16..23} vs
//                  {8..15,24..31}; 2 halfs per VGPR.
//   B (32x16 f16): lane&15 = col N; lane>>4 selects K {0..7,16..23} vs
//                  {8..15,24..31} (same K order as A).
//   C/D (16x16 f32): VGPR r -> M = r (lanes 0-15) / 8+r (lanes 16-31),
//                  N = lane&15.
// ---------------------------------------------------------------------------
__global__ __launch_bounds__(128)
void wmma_gemm_kernel(const _Float16* __restrict__ X,
                      const _Float16* __restrict__ Wt,
                      const float* __restrict__ bp,
                      _Float16* __restrict__ Y,
                      int M, int Kpad, int Npad, int relu)
{
  const int lane  = threadIdx.x & 31;
  const int wave  = threadIdx.x >> 5;
  const int mtile = blockIdx.x * 4 + wave;     // 32-row tile index
  if (mtile * 32 >= M) return;                 // wave-uniform: EXEC all-1s

  const int n0    = blockIdx.y * 32;
  const int arow0 = mtile * 32 + (lane & 15);
  const int arow1 = arow0 + 16;
  const int kA    = (lane >> 4) * 8;           // A: K sub-offset in 32-block
  const int kB    = (lane >> 4) * 16;          // unused (B loads also use kA)
  const int bcol0 = n0 + (lane & 15);
  const int bcol1 = bcol0 + 16;

  const _Float16* xr0 = X  + (size_t)arow0 * Kpad;
  const _Float16* xr1 = X  + (size_t)arow1 * Kpad;
  const _Float16* wr0 = Wt + (size_t)bcol0 * Kpad;
  const _Float16* wr1 = Wt + (size_t)bcol1 * Kpad;

  v8f acc00 = {};  // rows 0..15,  cols n0..n0+15
  v8f acc01 = {};  // rows 0..15,  cols n0+16..n0+31
  v8f acc10 = {};  // rows 16..31, cols n0..n0+15
  v8f acc11 = {};  // rows 16..31, cols n0+16..n0+31
  for (int kb = 0; kb < Kpad; kb += 32) {
    Frag16 a0, a1, b0, b1;
    a0.h[0] = *(const v8h*)(xr0 + kb + kA);
    a0.h[1] = *(const v8h*)(xr0 + kb + 16 + kA);
    a1.h[0] = *(const v8h*)(xr1 + kb + kA);
    a1.h[1] = *(const v8h*)(xr1 + kb + 16 + kA);
    b0.h[0] = *(const v8h*)(wr0 + kb + kA);
    b0.h[1] = *(const v8h*)(wr0 + kb + 16 + kA);
    b1.h[0] = *(const v8h*)(wr1 + kb + kA);
    b1.h[1] = *(const v8h*)(wr1 + kb + 16 + kA);
    acc00 = __builtin_amdgcn_wmma_f32_16x16x32_f16(false, a0.v, false, b0.v,
                                                   (short)0, acc00, false, false);
    acc01 = __builtin_amdgcn_wmma_f32_16x16x32_f16(false, a0.v, false, b1.v,
                                                   (short)0, acc01, false, false);
    acc10 = __builtin_amdgcn_wmma_f32_16x16x32_f16(false, a1.v, false, b0.v,
                                                   (short)0, acc10, false, false);
    acc11 = __builtin_amdgcn_wmma_f32_16x16x32_f16(false, a1.v, false, b1.v,
                                                   (short)0, acc11, false, false);
  }

  const int nlo = lane & 15;
  const int mhi = (lane >> 4) * 8;
  const float bias0 = bp[n0 + nlo];
  const float bias1 = bp[n0 + 16 + nlo];
#pragma unroll
  for (int r = 0; r < 8; ++r) {
    const int row0 = mtile * 32 + mhi + r;
    const int row1 = row0 + 16;
    float v00 = acc00[r] + bias0;
    float v01 = acc01[r] + bias1;
    float v10 = acc10[r] + bias0;
    float v11 = acc11[r] + bias1;
    if (relu) {
      v00 = v00 > 0.f ? v00 : 0.f;
      v01 = v01 > 0.f ? v01 : 0.f;
      v10 = v10 > 0.f ? v10 : 0.f;
      v11 = v11 > 0.f ? v11 : 0.f;
    }
    Y[(size_t)row0 * Npad + n0 + nlo]      = (_Float16)v00;
    Y[(size_t)row0 * Npad + n0 + 16 + nlo] = (_Float16)v01;
    Y[(size_t)row1 * Npad + n0 + nlo]      = (_Float16)v10;
    Y[(size_t)row1 * Npad + n0 + 16 + nlo] = (_Float16)v11;
  }
}

// --------------------------- weight preparation ----------------------------
// W [Cin,Cout] f32 row-major  ->  Wt [Npad][Kpad] f16 (transposed, zero pad)
__global__ void prep_wt_kernel(const float* __restrict__ W,
                               const float* __restrict__ bias,
                               _Float16* __restrict__ Wt,
                               float* __restrict__ bp,
                               int Cin, int Cout, int Kpad, int Npad)
{
  const int id = blockIdx.x * blockDim.x + threadIdx.x;
  if (id < Npad) bp[id] = (id < Cout) ? bias[id] : 0.f;
  const int total = Kpad * Npad;
  if (id >= total) return;
  const int k = id % Kpad;
  const int n = id / Kpad;
  Wt[id] = (k < Cin && n < Cout) ? (_Float16)W[(size_t)k * Cout + n]
                                 : (_Float16)0.f;
}

// ----------------------------- FPS (per batch) -----------------------------
__global__ __launch_bounds__(256)
void fps_kernel(const float* __restrict__ pos, int N, int npoint,
                int* __restrict__ out)
{
  const int b = blockIdx.x, tid = threadIdx.x;
  const float* P = pos + (size_t)b * N * 3;
  int* O = out + (size_t)b * npoint;
  __shared__ float sv[256];
  __shared__ int   si[256];
  float mind[16];
  const int cnt = (N + 255) >> 8;
  for (int r = 0; r < cnt; ++r) mind[r] = 1e10f;
  int last = 0;
  if (tid == 0) O[0] = 0;
  for (int i = 1; i < npoint; ++i) {
    const float lx = P[last*3+0], ly = P[last*3+1], lz = P[last*3+2];
    float bestv = -1.f; int besti = 0x7fffffff;
    for (int r = 0; r < cnt; ++r) {
      const int j = tid + (r << 8);
      if (j < N) {
        const float dx = P[j*3+0]-lx, dy = P[j*3+1]-ly, dz = P[j*3+2]-lz;
        const float d = dx*dx + dy*dy + dz*dz;
        const float m = fminf(mind[r], d);
        mind[r] = m;
        if (m > bestv) { bestv = m; besti = j; }   // strict >: lowest j on ties
      }
    }
    sv[tid] = bestv; si[tid] = besti;
    __syncthreads();
    for (int off = 128; off > 0; off >>= 1) {
      if (tid < off) {
        if (sv[tid+off] > sv[tid] ||
            (sv[tid+off] == sv[tid] && si[tid+off] < si[tid])) {
          sv[tid] = sv[tid+off]; si[tid] = si[tid+off];
        }
      }
      __syncthreads();
    }
    last = si[0];
    if (tid == 0) O[i] = last;
    __syncthreads();
  }
}

// ------------------------------ simple gathers -----------------------------
__global__ void gather_pos_kernel(const float* __restrict__ src,
                                  const int* __restrict__ idx,
                                  float* __restrict__ dst,
                                  int Nsrc, int S, int total)
{
  const int id = blockIdx.x * blockDim.x + threadIdx.x;
  if (id >= total) return;
  const int b = id / S;
  const int j = idx[id];
  const float* p = src + ((size_t)b * Nsrc + j) * 3;
  dst[(size_t)id*3+0] = p[0];
  dst[(size_t)id*3+1] = p[1];
  dst[(size_t)id*3+2] = p[2];
}

// ball query: first K in-radius point indices (ascending) per center
__global__ void ball_query_kernel(const float* __restrict__ centers,
                                  const float* __restrict__ pts,
                                  int S, int N, float r2, int K,
                                  int* __restrict__ nidx,
                                  int* __restrict__ cnt, int total)
{
  const int id = blockIdx.x * blockDim.x + threadIdx.x;
  if (id >= total) return;
  const int b = id / S;
  const float cx = centers[(size_t)id*3+0];
  const float cy = centers[(size_t)id*3+1];
  const float cz = centers[(size_t)id*3+2];
  const float* P = pts + (size_t)b * N * 3;
  int* out = nidx + (size_t)id * K;
  int c = 0;
  for (int j = 0; j < N && c < K; ++j) {
    const float dx = P[j*3+0]-cx, dy = P[j*3+1]-cy, dz = P[j*3+2]-cz;
    if (dx*dx + dy*dy + dz*dz < r2) out[c++] = j;
  }
  cnt[id] = c;
  for (int k = c; k < K; ++k) out[k] = 0;
}

// SA1 grouped input: [S1*K, 32] f16; cols 0..2 = rel pos, rest 0
__global__ void group_sa1_kernel(const float* __restrict__ pts,
                                 const float* __restrict__ centers,
                                 const int* __restrict__ nidx,
                                 const int* __restrict__ cnt,
                                 _Float16* __restrict__ gin, int K, int total)
{
  const int id = blockIdx.x * blockDim.x + threadIdx.x;
  if (id >= total) return;
  const int col = id & 31;
  const int row = id >> 5;
  const int s = row / K, k = row - s * K;
  _Float16 v = (_Float16)0.f;
  if (col < 3 && k < cnt[s]) {
    const int j = nidx[row];
    v = (_Float16)(pts[(size_t)j*3+col] - centers[(size_t)s*3+col]);
  }
  gin[(size_t)row * 32 + col] = v;
}

// SA2 grouped input: [S2*K, 352]; cols 0..319 = x1[neighbor], 320..322 = rel
__global__ void group_sa2_kernel(const _Float16* __restrict__ x1b,
                                 const float* __restrict__ pos1b,
                                 const float* __restrict__ centers2b,
                                 const int* __restrict__ nidx,
                                 const int* __restrict__ cnt,
                                 _Float16* __restrict__ gin, int K, int total)
{
  const int id = blockIdx.x * blockDim.x + threadIdx.x;
  if (id >= total) return;
  const int col = id % 352;
  const int row = id / 352;
  const int s = row / K, k = row - s * K;
  _Float16 v = (_Float16)0.f;
  if (k < cnt[s]) {
    const int j = nidx[row];
    if (col < 320) {
      v = x1b[(size_t)j * 320 + col];
    } else if (col < 323) {
      const int c = col - 320;
      v = (_Float16)(pos1b[(size_t)j*3+c] - centers2b[(size_t)s*3+c]);
    }
  }
  gin[(size_t)row * 352 + col] = v;
}

// masked max over K neighbors -> slice of concat feature tensor
__global__ void maxpool_kernel(const _Float16* __restrict__ g,
                               const int* __restrict__ cnt,
                               _Float16* __restrict__ xout,
                               int K, int Cpad, int Cout, int CTOT, int coff,
                               int total)
{
  const int id = blockIdx.x * blockDim.x + threadIdx.x;
  if (id >= total) return;
  const int c = id % Cout;
  const int s = id / Cout;
  const int n = cnt[s];                       // >= 1 (center in own ball)
  float m = (float)g[(size_t)(s * K) * Cpad + c];
  for (int k = 1; k < n; ++k)
    m = fmaxf(m, (float)g[(size_t)(s * K + k) * Cpad + c]);
  xout[(size_t)s * CTOT + coff + c] = (_Float16)m;
}

// SA3 input: [B*S2, 672] = concat(x2[640], pos2[3], pad)
__global__ void concat_sa3_kernel(const _Float16* __restrict__ x2,
                                  const float* __restrict__ pos2,
                                  _Float16* __restrict__ out, int total)
{
  const int id = blockIdx.x * blockDim.x + threadIdx.x;
  if (id >= total) return;
  const int col = id % 672;
  const int row = id / 672;
  _Float16 v = (_Float16)0.f;
  if (col < 640)      v = x2[(size_t)row * 640 + col];
  else if (col < 643) v = (_Float16)pos2[(size_t)row * 3 + (col - 640)];
  out[(size_t)row * 672 + col] = v;
}

// per-batch max over rows (SA3 global pool)
__global__ void rowmax_kernel(const _Float16* __restrict__ h,
                              _Float16* __restrict__ out,
                              int rows, int C, int total)
{
  const int id = blockIdx.x * blockDim.x + threadIdx.x;
  if (id >= total) return;
  const int c = id % C;
  const int b = id / C;
  float m = (float)h[(size_t)(b * rows) * C + c];
  for (int k = 1; k < rows; ++k)
    m = fmaxf(m, (float)h[(size_t)(b * rows + k) * C + c]);
  out[id] = (_Float16)m;
}

// FP3 input: [B*S2, 1664] = concat(broadcast x3[1024], x2[640])
__global__ void concat_fp3_kernel(const _Float16* __restrict__ x3,
                                  const _Float16* __restrict__ x2,
                                  _Float16* __restrict__ out, int total)
{
  const int id = blockIdx.x * blockDim.x + threadIdx.x;
  if (id >= total) return;
  const int col = id % 1664;
  const int row = id / 1664;
  const int b = row >> 7;                       // S2 == 128
  const _Float16 v = (col < 1024) ? x3[(size_t)b * 1024 + col]
                                  : x2[(size_t)row * 640 + (col - 1024)];
  out[(size_t)row * 1664 + col] = v;
}

// 3-NN with inverse-squared-distance weights (ties -> lowest index)
__global__ void knn3_kernel(const float* __restrict__ dst,
                            const float* __restrict__ src,
                            int Nt, int Ns,
                            int* __restrict__ idx, float* __restrict__ w,
                            int total)
{
  const int id = blockIdx.x * blockDim.x + threadIdx.x;
  if (id >= total) return;
  const int b = id / Nt;
  const float tx = dst[(size_t)id*3+0];
  const float ty = dst[(size_t)id*3+1];
  const float tz = dst[(size_t)id*3+2];
  const float* S = src + (size_t)b * Ns * 3;
  float d0 = 3.4e38f, d1 = 3.4e38f, d2 = 3.4e38f;
  int   i0 = 0, i1 = 0, i2 = 0;
  for (int j = 0; j < Ns; ++j) {
    const float dx = S[j*3+0]-tx, dy = S[j*3+1]-ty, dz = S[j*3+2]-tz;
    const float d = dx*dx + dy*dy + dz*dz;
    if (d < d0)      { d2 = d1; i2 = i1; d1 = d0; i1 = i0; d0 = d; i0 = j; }
    else if (d < d1) { d2 = d1; i2 = i1; d1 = d;  i1 = j; }
    else if (d < d2) { d2 = d;  i2 = j; }
  }
  const float w0 = 1.f / fmaxf(d0, 1e-16f);
  const float w1 = 1.f / fmaxf(d1, 1e-16f);
  const float w2 = 1.f / fmaxf(d2, 1e-16f);
  const float sum = w0 + w1 + w2;
  idx[(size_t)id*3+0] = i0; idx[(size_t)id*3+1] = i1; idx[(size_t)id*3+2] = i2;
  w[(size_t)id*3+0] = w0/sum; w[(size_t)id*3+1] = w1/sum; w[(size_t)id*3+2] = w2/sum;
}

// interpolate (3-NN weighted sum of fsrc) and optionally concat xcat
__global__ void interp_concat_kernel(const _Float16* __restrict__ fsrc,
                                     int Ns, int Cs,
                                     const int* __restrict__ idx,
                                     const float* __restrict__ w,
                                     const _Float16* __restrict__ xcat, int Cc,
                                     _Float16* __restrict__ out,
                                     int Nt, int Ctot, int total)
{
  const int id = blockIdx.x * blockDim.x + threadIdx.x;
  if (id >= total) return;
  const int col = id % Ctot;
  const int row = id / Ctot;
  const int b = row / Nt;
  _Float16 v;
  if (col < Cs) {
    const int*   I  = idx + (size_t)row * 3;
    const float* W_ = w   + (size_t)row * 3;
    float acc = 0.f;
    for (int q = 0; q < 3; ++q)
      acc += W_[q] * (float)fsrc[((size_t)b * Ns + I[q]) * Cs + col];
    v = (_Float16)acc;
  } else {
    v = xcat[(size_t)row * Cc + (col - Cs)];
  }
  out[(size_t)row * Ctot + col] = v;
}

// log-softmax over 13 classes (logits stored in [*,32] padded f16)
__global__ void log_softmax_kernel(const _Float16* __restrict__ logits,
                                   float* __restrict__ out, int total)
{
  const int id = blockIdx.x * blockDim.x + threadIdx.x;
  if (id >= total) return;
  const _Float16* L = logits + (size_t)id * 32;
  float x[13];
  float m = -3.4e38f;
  for (int c = 0; c < 13; ++c) { x[c] = (float)L[c]; m = fmaxf(m, x[c]); }
  float s = 0.f;
  for (int c = 0; c < 13; ++c) s += expf(x[c] - m);
  const float lse = logf(s);
  for (int c = 0; c < 13; ++c) out[(size_t)id * 13 + c] = x[c] - m - lse;
}

// ---------------------------------------------------------------------------
// Host orchestration
// ---------------------------------------------------------------------------
extern "C" void kernel_launch(void* const* d_in, const int* in_sizes, int n_in,
                              void* d_out, int out_size, void* d_ws, size_t ws_size,
                              hipStream_t stream)
{
  (void)in_sizes; (void)n_in; (void)out_size; (void)ws_size;

  // Input ordering assumption: d_in[0] = pos, then params leaves in dict
  // insertion order: sa1(3 MLPs x 3 layers), sa2, sa3, fp3, fp2, fp1,
  // lin1, lin2, lin3; each layer contributes (W, b).
  const float* pos = (const float*)d_in[0];

  // bump allocator over d_ws
  size_t off = 0;
  auto alloc = [&](size_t bytes) -> void* {
    off = (off + 255) & ~(size_t)255;
    void* p = (char*)d_ws + off;
    off += bytes;
    return p;
  };

  struct Lyr { int wi, Cin, Cout; int Kpad, Npad; _Float16* Wt; float* bp; };
  Lyr L[31] = {
    // sa1: [3,32,32,64], [3,64,64,128], [3,64,96,128]
    {1,3,32},{3,32,32},{5,32,64},
    {7,3,64},{9,64,64},{11,64,128},
    {13,3,64},{15,64,96},{17,96,128},
    // sa2: [323,64,64,128], [323,128,128,256], [323,128,128,256]
    {19,323,64},{21,64,64},{23,64,128},
    {25,323,128},{27,128,128},{29,128,256},
    {31,323,128},{33,128,128},{35,128,256},
    // sa3: [643,256,512,1024]
    {37,643,256},{39,256,512},{41,512,1024},
    // fp3: [1664,256,256]
    {43,1664,256},{45,256,256},
    // fp2: [576,256,128]
    {47,576,256},{49,256,128},
    // fp1: [128,128,128,128]
    {51,128,128},{53,128,128},{55,128,128},
    // lin1, lin2, lin3
    {57,128,128},{59,128,128},{61,128,13}
  };
  for (int i = 0; i < 31; ++i) {
    L[i].Kpad = ((L[i].Cin  + 31) / 32) * 32;
    L[i].Npad = ((L[i].Cout + 31) / 32) * 32;
    L[i].Wt = (_Float16*)alloc((size_t)L[i].Kpad * L[i].Npad * sizeof(_Float16));
    L[i].bp = (float*)alloc((size_t)L[i].Npad * sizeof(float));
  }

  // persistent activations / indices
  int*   idx1 = (int*)  alloc((size_t)BATCH*SS1*sizeof(int));
  float* pos1 = (float*)alloc((size_t)BATCH*SS1*3*sizeof(float));
  int*   idx2 = (int*)  alloc((size_t)BATCH*SS2*sizeof(int));
  float* pos2 = (float*)alloc((size_t)BATCH*SS2*3*sizeof(float));

  const int   K1[3]   = {16, 32, 128};
  const float R1sq[3] = {0.01f, 0.04f, 0.16f};
  const int   K2[3]   = {32, 64, 128};
  const float R2sq[3] = {0.04f, 0.16f, 0.64f};

  int* nidx1[3]; int* cnt1[3];
  for (int r = 0; r < 3; ++r) {
    nidx1[r] = (int*)alloc((size_t)BATCH*SS1*K1[r]*sizeof(int));
    cnt1[r]  = (int*)alloc((size_t)BATCH*SS1*sizeof(int));
  }
  int* nidx2[3]; int* cnt2[3];
  for (int r = 0; r < 3; ++r) {
    nidx2[r] = (int*)alloc((size_t)BATCH*SS2*K2[r]*sizeof(int));
    cnt2[r]  = (int*)alloc((size_t)BATCH*SS2*sizeof(int));
  }

  _Float16* x1 = (_Float16*)alloc((size_t)BATCH*SS1*320*sizeof(_Float16));
  _Float16* x2 = (_Float16*)alloc((size_t)BATCH*SS2*640*sizeof(_Float16));
  _Float16* x3 = (_Float16*)alloc((size_t)BATCH*1024*sizeof(_Float16));
  _Float16* f3 = (_Float16*)alloc((size_t)BATCH*SS2*256*sizeof(_Float16));
  _Float16* f2 = (_Float16*)alloc((size_t)BATCH*SS1*128*sizeof(_Float16));
  int*   kidx2 = (int*)  alloc((size_t)BATCH*SS1*3*sizeof(int));
  float* kw2   = (float*)alloc((size_t)BATCH*SS1*3*sizeof(float));
  int*   kidx1 = (int*)  alloc((size_t)BATCH*NPTS*3*sizeof(int));
  float* kw1   = (float*)alloc((size_t)BATCH*NPTS*3*sizeof(float));

  // scratch GEMM buffers (per-batch chunking for grouped SA stages)
  const size_t GIN_HALFS = (size_t)SS2 * 128 * 352;   // 5,767,168 (>= 65536*32, >= 32768*128)
  const size_t PP_HALFS  = (size_t)SS1 * 128 * 128;   // 8,388,608
  _Float16* gin = (_Float16*)alloc(GIN_HALFS * sizeof(_Float16));
  _Float16* pp0 = (_Float16*)alloc(PP_HALFS  * sizeof(_Float16));
  _Float16* pp1 = (_Float16*)alloc(PP_HALFS  * sizeof(_Float16));

  auto gemm = [&](const _Float16* X, const Lyr& Ly, _Float16* Y, int M, int relu) {
    dim3 grid((unsigned)((M + 127) / 128), (unsigned)(Ly.Npad / 32));
    wmma_gemm_kernel<<<grid, 128, 0, stream>>>(X, Ly.Wt, Ly.bp, Y,
                                               M, Ly.Kpad, Ly.Npad, relu);
  };

  // ---- weight preparation ----
  for (int i = 0; i < 31; ++i) {
    const int total = L[i].Kpad * L[i].Npad;
    prep_wt_kernel<<<(total + 255) / 256, 256, 0, stream>>>(
        (const float*)d_in[L[i].wi], (const float*)d_in[L[i].wi + 1],
        L[i].Wt, L[i].bp, L[i].Cin, L[i].Cout, L[i].Kpad, L[i].Npad);
  }

  // ---- SA1: FPS + ball query ----
  fps_kernel<<<BATCH, 256, 0, stream>>>(pos, NPTS, SS1, idx1);
  int tot = BATCH * SS1;
  gather_pos_kernel<<<(tot + 255) / 256, 256, 0, stream>>>(pos, idx1, pos1, NPTS, SS1, tot);
  for (int r = 0; r < 3; ++r)
    ball_query_kernel<<<(tot + 255) / 256, 256, 0, stream>>>(
        pos1, pos, SS1, NPTS, R1sq[r], K1[r], nidx1[r], cnt1[r], tot);

  // ---- SA1 grouped MLPs (per batch, per scale) ----
  const int coff1[3] = {0, 64, 192};
  for (int b = 0; b < BATCH; ++b) {
    for (int r = 0; r < 3; ++r) {
      const int K = K1[r], rows = SS1 * K;
      const int tg = rows * 32;
      group_sa1_kernel<<<(tg + 255) / 256, 256, 0, stream>>>(
          pos + (size_t)b * NPTS * 3, pos1 + (size_t)b * SS1 * 3,
          nidx1[r] + (size_t)b * SS1 * K, cnt1[r] + b * SS1, gin, K, tg);
      const Lyr* C = &L[r * 3];
      gemm(gin, C[0], pp0, rows, 1);
      gemm(pp0, C[1], pp1, rows, 1);
      gemm(pp1, C[2], pp0, rows, 1);
      const int tp = SS1 * C[2].Cout;
      maxpool_kernel<<<(tp + 255) / 256, 256, 0, stream>>>(
          pp0, cnt1[r] + b * SS1, x1 + (size_t)b * SS1 * 320,
          K, C[2].Npad, C[2].Cout, 320, coff1[r], tp);
    }
  }

  // ---- SA2: FPS + ball query ----
  fps_kernel<<<BATCH, 256, 0, stream>>>(pos1, SS1, SS2, idx2);
  tot = BATCH * SS2;
  gather_pos_kernel<<<(tot + 255) / 256, 256, 0, stream>>>(pos1, idx2, pos2, SS1, SS2, tot);
  for (int r = 0; r < 3; ++r)
    ball_query_kernel<<<(tot + 255) / 256, 256, 0, stream>>>(
        pos2, pos1, SS2, SS1, R2sq[r], K2[r], nidx2[r], cnt2[r], tot);

  // ---- SA2 grouped MLPs ----
  const int coff2[3] = {0, 128, 384};
  for (int b = 0; b < BATCH; ++b) {
    for (int r = 0; r < 3; ++r) {
      const int K = K2[r], rows = SS2 * K;
      const int tg = rows * 352;
      group_sa2_kernel<<<(tg + 255) / 256, 256, 0, stream>>>(
          x1 + (size_t)b * SS1 * 320, pos1 + (size_t)b * SS1 * 3,
          pos2 + (size_t)b * SS2 * 3,
          nidx2[r] + (size_t)b * SS2 * K, cnt2[r] + b * SS2, gin, K, tg);
      const Lyr* C = &L[9 + r * 3];
      gemm(gin, C[0], pp0, rows, 1);
      gemm(pp0, C[1], pp1, rows, 1);
      gemm(pp1, C[2], pp0, rows, 1);
      const int tp = SS2 * C[2].Cout;
      maxpool_kernel<<<(tp + 255) / 256, 256, 0, stream>>>(
          pp0, cnt2[r] + b * SS2, x2 + (size_t)b * SS2 * 640,
          K, C[2].Npad, C[2].Cout, 640, coff2[r], tp);
    }
  }

  // ---- SA3: concat + MLP + global max ----
  const int M3 = BATCH * SS2;                       // 1024
  tot = M3 * 672;
  concat_sa3_kernel<<<(tot + 255) / 256, 256, 0, stream>>>(x2, pos2, gin, tot);
  gemm(gin, L[18], pp0, M3, 1);
  gemm(pp0, L[19], pp1, M3, 1);
  gemm(pp1, L[20], pp0, M3, 1);
  tot = BATCH * 1024;
  rowmax_kernel<<<(tot + 255) / 256, 256, 0, stream>>>(pp0, x3, SS2, 1024, tot);

  // ---- FP3: broadcast x3 + concat x2, MLP ----
  tot = M3 * 1664;
  concat_fp3_kernel<<<(tot + 255) / 256, 256, 0, stream>>>(x3, x2, gin, tot);
  gemm(gin, L[21], pp0, M3, 1);
  gemm(pp0, L[22], f3, M3, 1);

  // ---- FP2: 3-NN interp(f3, pos2 -> pos1) ++ x1, MLP ----
  tot = BATCH * SS1;
  knn3_kernel<<<(tot + 255) / 256, 256, 0, stream>>>(pos1, pos2, SS1, SS2, kidx2, kw2, tot);
  tot = BATCH * SS1 * 576;
  interp_concat_kernel<<<(tot + 255) / 256, 256, 0, stream>>>(
      f3, SS2, 256, kidx2, kw2, x1, 320, gin, SS1, 576, tot);
  const int M2 = BATCH * SS1;                       // 4096
  gemm(gin, L[23], pp0, M2, 1);
  gemm(pp0, L[24], f2, M2, 1);

  // ---- FP1: 3-NN interp(f2, pos1 -> pos), MLP ----
  tot = BATCH * NPTS;
  knn3_kernel<<<(tot + 255) / 256, 256, 0, stream>>>(pos, pos1, NPTS, SS1, kidx1, kw1, tot);
  tot = BATCH * NPTS * 128;
  interp_concat_kernel<<<(tot + 255) / 256, 256, 0, stream>>>(
      f2, SS1, 128, kidx1, kw1, (const _Float16*)nullptr, 0, gin, NPTS, 128, tot);
  const int M1 = BATCH * NPTS;                      // 32768
  gemm(gin, L[25], pp0, M1, 1);
  gemm(pp0, L[26], pp1, M1, 1);
  gemm(pp1, L[27], pp0, M1, 1);

  // ---- heads: lin1(+relu), lin2, lin3, log_softmax ----
  gemm(pp0, L[28], pp1, M1, 1);
  gemm(pp1, L[29], gin, M1, 0);
  gemm(gin, L[30], pp0, M1, 0);                     // logits [32768, 32]
  tot = M1;
  log_softmax_kernel<<<(tot + 255) / 256, 256, 0, stream>>>(pp0, (float*)d_out, tot);
}
